// AttentionBlockV2_65180423685286
// MI455X (gfx1250) — matrix operand order repaired and checked by reading, not compiled, not run on hardware
//
#include <hip/hip_runtime.h>
#include <hip/hip_bf16.h>

typedef __attribute__((ext_vector_type(16))) _Float16 v16h;
typedef __attribute__((ext_vector_type(8)))  _Float16 v8h;
typedef __attribute__((ext_vector_type(8)))  float    v8f;

#define CAT16(lo, hi) __builtin_shufflevector((lo), (hi), 0,1,2,3,4,5,6,7,8,9,10,11,12,13,14,15)

// LDS byte offset of a generic pointer into shared memory (for async-LDS asm).
__device__ inline unsigned lds_addr(const void* p) {
  return (unsigned)(unsigned long long)(__attribute__((address_space(3))) const char*)p;
}

// ---------------------------------------------------------------------------
// f32 -> f16 convert (same layout)
// ---------------------------------------------------------------------------
__global__ void k_f32_to_f16(const float* __restrict__ s, _Float16* __restrict__ d, int n) {
  int i = blockIdx.x * blockDim.x + threadIdx.x;
  if (i < n) d[i] = (_Float16)s[i];
}

// f32 (b,c,n) -> f16 (b,n,c)  [k-major activations for WMMA operand loads]
__global__ void k_f32_to_f16_t(const float* __restrict__ s, _Float16* __restrict__ d,
                               int C_, int Nn_, int total) {
  int i = blockIdx.x * blockDim.x + threadIdx.x;
  if (i >= total) return;
  int n = i % Nn_;
  int c = (i / Nn_) % C_;
  int b = i / (Nn_ * C_);
  d[((size_t)b * Nn_ + n) * C_ + c] = (_Float16)s[i];
}

// d(b,n,c) = f16(a(b,c,n) + bb(b,c,n))
__global__ void k_add_to_f16_t(const float* __restrict__ a, const float* __restrict__ bb,
                               _Float16* __restrict__ d, int C_, int Nn_, int total) {
  int i = blockIdx.x * blockDim.x + threadIdx.x;
  if (i >= total) return;
  int n = i % Nn_;
  int c = (i / Nn_) % C_;
  int b = i / (Nn_ * C_);
  d[((size_t)b * Nn_ + n) * C_ + c] = (_Float16)(a[i] + bb[i]);
}

// ---------------------------------------------------------------------------
// Depthwise 7x7 conv, SAME padding: pe = dwconv(v_full, pe_w) + pe_b
// ---------------------------------------------------------------------------
__global__ void k_dwconv7(const float* __restrict__ v, const float* __restrict__ w,
                          const float* __restrict__ bias, float* __restrict__ pe,
                          int Bn, int C, int H, int W) {
  int idx = blockIdx.x * blockDim.x + threadIdx.x;
  int total = Bn * C * H * W;
  if (idx >= total) return;
  int x  = idx % W;
  int y  = (idx / W) % H;
  int c  = (idx / (W * H)) % C;
  int bb = idx / (W * H * C);
  const float* vp = v + ((size_t)bb * C + c) * H * W;
  const float* wp = w + c * 49;
  float s = bias[c];
#pragma unroll
  for (int ky = 0; ky < 7; ++ky) {
    int yy = y + ky - 3;
    if (yy < 0 || yy >= H) continue;
#pragma unroll
    for (int kx = 0; kx < 7; ++kx) {
      int xx = x + kx - 3;
      if (xx < 0 || xx >= W) continue;
      s += vp[yy * W + xx] * wp[ky * 7 + kx];
    }
  }
  pe[idx] = s;
}

// ---------------------------------------------------------------------------
// WMMA GEMM: out[b,m,n] = act( sum_k Wm[m,k]*X[b,n,k] + bias[m] (+res) )
// X is k-major (b, Nn, K). Block = 256 thr (8 waves), tile 64(M) x 128(N).
// Each wave: 32x32 output = 2x2 WMMA tiles, 4 v_wmma per 32-K step.
// Staging uses CDNA5 async global->LDS (ASYNCcnt), fragments are b128 LDS loads.
// ---------------------------------------------------------------------------
__global__ __launch_bounds__(256) void k_gemm(
    const _Float16* __restrict__ Wm,   // (M, K) row-major
    const _Float16* __restrict__ X,    // (b, Nn, K)
    const float* __restrict__ bias,    // (M)
    const float* __restrict__ res,     // (b, M, Nn) f32 or null
    float* __restrict__ outF,          // (b, M, Nn) f32 or null
    _Float16* __restrict__ outH,       // f16 out or null
    int M, int Nn, int K, int act, int hTrans) {  // hTrans: 1 -> outH (b,Nn,M); 0 -> (b,M,Nn)
  __shared__ _Float16 As[64][40];    // 64 m-rows x 32 k (pitch 80B, 16B aligned)
  __shared__ _Float16 Bs[128][40];   // 128 n-rows x 32 k

  const int b    = blockIdx.z;
  const int n0   = blockIdx.x * 128;
  const int m0   = blockIdx.y * 64;
  const int t    = threadIdx.x;
  const int lane = t & 31;
  const int wave = t >> 5;
  const int wm   = (wave >> 2) * 32;  // 0 | 32
  const int wn   = (wave & 3) * 32;   // 0 | 32 | 64 | 96
  const int col  = lane & 15;
  const int half = lane >> 4;         // 0 | 1
  const int kh   = half * 8;          // A-frag K-half
  const int kh16 = half * 16;         // unused; B-frag also uses kh
  const int rbase = half * 8;

  const _Float16* Xb = X + (size_t)b * Nn * K;

  // staging: A 64x32 -> 8 f16/thread ; B 128x32 -> 16 f16/thread (2x16B)
  const int am = t >> 2, ak = (t & 3) * 8;
  const int bn = t >> 1, bk = (t & 1) * 16;
  const unsigned ldsA = lds_addr(&As[am][ak]);
  const unsigned ldsB = lds_addr(&Bs[bn][bk]);

  v8f acc[2][2] = {};

  for (int k0 = 0; k0 < K; k0 += 32) {
    const _Float16* gA = Wm + (size_t)(m0 + am) * K + k0 + ak;
    const _Float16* gB = Xb + (size_t)(n0 + bn) * K + k0 + bk;
    // CDNA5 async global->LDS copies (offset applies to both LDS and global addr)
    asm volatile("global_load_async_to_lds_b128 %0, %1, off"
                 :: "v"(ldsA), "v"(gA) : "memory");
    asm volatile("global_load_async_to_lds_b128 %0, %1, off"
                 :: "v"(ldsB), "v"(gB) : "memory");
    asm volatile("global_load_async_to_lds_b128 %0, %1, off offset:16"
                 :: "v"(ldsB), "v"(gB) : "memory");
    asm volatile("s_wait_asynccnt 0x0" ::: "memory");
    __syncthreads();

    // fragments: all contiguous 16B LDS loads
    const _Float16* a0 = &As[wm + col][kh];
    const _Float16* a1 = &As[wm + 16 + col][kh];
    const _Float16* b0 = &Bs[wn + col][kh];
    const _Float16* b1 = &Bs[wn + 16 + col][kh];
    v16h af0 = CAT16(*(const v8h*)a0, *(const v8h*)(a0 + 16));
    v16h af1 = CAT16(*(const v8h*)a1, *(const v8h*)(a1 + 16));
    v16h bf0 = CAT16(*(const v8h*)b0, *(const v8h*)(b0 + 16));
    v16h bf1 = CAT16(*(const v8h*)b1, *(const v8h*)(b1 + 16));

    acc[0][0] = __builtin_amdgcn_wmma_f32_16x16x32_f16(false, af0, false, bf0, (short)0, acc[0][0], false, false);
    acc[0][1] = __builtin_amdgcn_wmma_f32_16x16x32_f16(false, af0, false, bf1, (short)0, acc[0][1], false, false);
    acc[1][0] = __builtin_amdgcn_wmma_f32_16x16x32_f16(false, af1, false, bf0, (short)0, acc[1][0], false, false);
    acc[1][1] = __builtin_amdgcn_wmma_f32_16x16x32_f16(false, af1, false, bf1, (short)0, acc[1][1], false, false);
    __syncthreads();
  }

  // epilogue: C/D layout -> VGPR r = row (rbase+r), lane col = n
#pragma unroll
  for (int i2 = 0; i2 < 2; ++i2) {
#pragma unroll
    for (int j2 = 0; j2 < 2; ++j2) {
      int gmb = m0 + wm + i2 * 16 + rbase;
      int gn  = n0 + wn + j2 * 16 + col;
      float vals[8];
#pragma unroll
      for (int r = 0; r < 8; ++r) {
        float v = acc[i2][j2][r] + bias[gmb + r];
        size_t oidx = ((size_t)b * M + gmb + r) * Nn + gn;
        if (res) v += res[oidx];
        if (act == 1) v = v / (1.0f + __expf(-v));   // SiLU
        if (outF) outF[oidx] = v;
        vals[r] = v;
      }
      if (outH) {
        if (hTrans) {          // (b, n, m): 8 consecutive m -> one 16B store
          v8h hv;
#pragma unroll
          for (int r = 0; r < 8; ++r) hv[r] = (_Float16)vals[r];
          *(v8h*)(outH + ((size_t)b * Nn + gn) * M + gmb) = hv;
        } else {               // (b, m, n)
#pragma unroll
          for (int r = 0; r < 8; ++r)
            outH[((size_t)b * M + gmb + r) * Nn + gn] = (_Float16)vals[r];
        }
      }
    }
  }
}

// ---------------------------------------------------------------------------
// Flash attention, one wave per (b, head, 16-query tile). head_dim = 32.
// qkh: (B, N, 512) k-major f16 (cols 0..255 = q, 256..511 = k channels)
// vh : (B, 256, N) m-major f16 ; obuf: (B, 256, N) f32
// ---------------------------------------------------------------------------
__global__ __launch_bounds__(32) void k_flash(
    const _Float16* __restrict__ qkh, const _Float16* __restrict__ vh,
    float* __restrict__ obuf, int Nn) {
  __shared__ _Float16 Ps[16][40];   // P tile 16(n) x 32(m), pitch 80B

  const int b    = blockIdx.z;
  const int h    = blockIdx.y;
  const int n0   = blockIdx.x * 16;
  const int lane = threadIdx.x & 31;
  const int col  = lane & 15;
  const int half = lane >> 4;
  const int kh    = half * 8;
  const int kh16  = half * 16;
  const int rbase = half * 8;

  const _Float16* qn = qkh + (size_t)b * Nn * 512;        // (n, 512)
  const _Float16* kp = qn + 256 + h * 32;                 // k channels
  const _Float16* vb = vh + ((size_t)b * 256 + h * 32) * Nn;

  const float scale = 0.17677669529663687f;  // 1/sqrt(32)

  // Q^T fragment (A 16x32): row = query n, K = head dim (contiguous in memory)
  v16h aq;
  {
    const _Float16* qrow = qn + (size_t)(n0 + col) * 512 + h * 32;
    v8h lo = *(const v8h*)(qrow + kh);
    v8h hi = *(const v8h*)(qrow + 16 + kh);
#pragma unroll
    for (int i = 0; i < 8; ++i) aq[i]     = (_Float16)((float)lo[i] * scale);
#pragma unroll
    for (int i = 0; i < 8; ++i) aq[8 + i] = (_Float16)((float)hi[i] * scale);
  }

  float mrow[8], lrow[8];
#pragma unroll
  for (int r = 0; r < 8; ++r) { mrow[r] = -1e30f; lrow[r] = 0.0f; }
  v8f o0 = {}, o1 = {};   // d = 0..15 / 16..31 accumulators

  for (int m0 = 0; m0 < Nn; m0 += 32) {
    // K fragments (B 32x16, K = head dim): contiguous b128 loads per lane
    const _Float16* k0p = kp + (size_t)(m0 + col) * 512 + kh;
    const _Float16* k1p = kp + (size_t)(m0 + 16 + col) * 512 + kh;
    v16h kf0 = CAT16(*(const v8h*)k0p, *(const v8h*)(k0p + 16));
    v16h kf1 = CAT16(*(const v8h*)k1p, *(const v8h*)(k1p + 16));

    v8f z = {};
    v8f s0 = __builtin_amdgcn_wmma_f32_16x16x32_f16(false, aq, false, kf0, (short)0, z, false, false);
    v8f s1 = __builtin_amdgcn_wmma_f32_16x16x32_f16(false, aq, false, kf1, (short)0, z, false, false);

    // online softmax; row n = rbase+r lives in VGPR r across each 16-lane group
#pragma unroll
    for (int r = 0; r < 8; ++r) {
      float mo = mrow[r];
      float sm = fmaxf(s0[r], s1[r]);
#pragma unroll
      for (int msk = 1; msk < 16; msk <<= 1) sm = fmaxf(sm, __shfl_xor(sm, msk, 32));
      float mn   = fmaxf(mo, sm);
      float p0   = __expf(s0[r] - mn);
      float p1   = __expf(s1[r] - mn);
      float beta = __expf(mo - mn);
      float rs   = p0 + p1;
#pragma unroll
      for (int msk = 1; msk < 16; msk <<= 1) rs += __shfl_xor(rs, msk, 32);
      lrow[r] = lrow[r] * beta + rs;
      mrow[r] = mn;
      o0[r] *= beta;
      o1[r] *= beta;
      Ps[rbase + r][col]      = (_Float16)p0;
      Ps[rbase + r][16 + col] = (_Float16)p1;
    }
    __syncthreads();

    // P as A fragment (16 n x 32 m): contiguous b128 LDS loads
    const _Float16* prow = &Ps[col][0];
    v16h ap = CAT16(*(const v8h*)(prow + kh), *(const v8h*)(prow + 16 + kh));

    // V^T fragments (B 32x16, K = key index m): contiguous in m-major vh
    const _Float16* v0p = vb + (size_t)col * Nn + m0 + kh;
    const _Float16* v1p = vb + (size_t)(16 + col) * Nn + m0 + kh;
    v16h vf0 = CAT16(*(const v8h*)v0p, *(const v8h*)(v0p + 16));
    v16h vf1 = CAT16(*(const v8h*)v1p, *(const v8h*)(v1p + 16));

    o0 = __builtin_amdgcn_wmma_f32_16x16x32_f16(false, ap, false, vf0, (short)0, o0, false, false);
    o1 = __builtin_amdgcn_wmma_f32_16x16x32_f16(false, ap, false, vf1, (short)0, o1, false, false);
    __syncthreads();
  }

#pragma unroll
  for (int r = 0; r < 8; ++r) {
    float inv = 1.0f / lrow[r];
    size_t n = (size_t)(n0 + rbase + r);
    obuf[((size_t)b * 256 + h * 32 + col) * Nn + n]      = o0[r] * inv;
    obuf[((size_t)b * 256 + h * 32 + 16 + col) * Nn + n] = o1[r] * inv;
  }
}

// ---------------------------------------------------------------------------
extern "C" void kernel_launch(void* const* d_in, const int* in_sizes, int n_in,
                              void* d_out, int out_size, void* d_ws, size_t ws_size,
                              hipStream_t stream) {
  const float* x      = (const float*)d_in[0];
  const float* qk_w   = (const float*)d_in[1];
  const float* qk_b   = (const float*)d_in[2];
  const float* v_w    = (const float*)d_in[3];
  const float* v_b    = (const float*)d_in[4];
  const float* pe_w   = (const float*)d_in[5];
  const float* pe_b   = (const float*)d_in[6];
  const float* proj_w = (const float*)d_in[7];
  const float* proj_b = (const float*)d_in[8];
  const float* fc1_w  = (const float*)d_in[9];
  const float* fc1_b  = (const float*)d_in[10];
  const float* fc2_w  = (const float*)d_in[11];
  const float* fc2_b  = (const float*)d_in[12];
  float* out = (float*)d_out;

  const int Bn = 2, C = 256, H = 48, Wd = 48, NH = 8;
  const int Nn = H * Wd;            // 2304
  const int BCN = Bn * C * Nn;      // 1,179,648

  // ---- carve workspace (256B aligned), lifetime-based reuse ----
  char* p = (char*)d_ws;
  auto carve = [&](size_t bytes) -> void* {
    void* r = (void*)p;
    p += (bytes + 255) & ~(size_t)255;
    return r;
  };
  _Float16* xh    = (_Float16*)carve((size_t)BCN * 2);           // x f16 (b,n,c); reused as oph
  _Float16* wqk   = (_Float16*)carve((size_t)512 * 256 * 2);
  _Float16* wv    = (_Float16*)carve((size_t)256 * 256 * 2);
  _Float16* wproj = (_Float16*)carve((size_t)256 * 256 * 2);
  _Float16* wfc1  = (_Float16*)carve((size_t)512 * 256 * 2);
  _Float16* wfc2  = (_Float16*)carve((size_t)256 * 512 * 2);
  _Float16* qkh   = (_Float16*)carve((size_t)Bn * 512 * Nn * 2); // (b,n,512); reused as hbuf
  float*    vfull = (float*)   carve((size_t)BCN * 4);           // (b,c,n); reused as x1
  _Float16* vhh   = (_Float16*)carve((size_t)BCN * 2);           // (b,c,n) m-major for V frags
  float*    pe    = (float*)   carve((size_t)BCN * 4);
  float*    obuf  = (float*)   carve((size_t)BCN * 4);
  _Float16* x1h   = (_Float16*)carve((size_t)BCN * 2);           // (b,n,c)
  _Float16* oph   = xh;       // x f16 dead after qk/v GEMMs
  float*    x1    = vfull;    // v_full dead after dwconv
  _Float16* hbuf  = qkh;      // qk f16 dead after flash attention

  auto cdiv = [](int a, int b) { return (a + b - 1) / b; };

  // ---- converts ----
  k_f32_to_f16_t<<<cdiv(BCN, 256), 256, 0, stream>>>(x, xh, C, Nn, BCN);
  k_f32_to_f16<<<cdiv(512 * 256, 256), 256, 0, stream>>>(qk_w, wqk, 512 * 256);
  k_f32_to_f16<<<cdiv(256 * 256, 256), 256, 0, stream>>>(v_w, wv, 256 * 256);
  k_f32_to_f16<<<cdiv(256 * 256, 256), 256, 0, stream>>>(proj_w, wproj, 256 * 256);
  k_f32_to_f16<<<cdiv(512 * 256, 256), 256, 0, stream>>>(fc1_w, wfc1, 512 * 256);
  k_f32_to_f16<<<cdiv(256 * 512, 256), 256, 0, stream>>>(fc2_w, wfc2, 256 * 512);

  // ---- qk = W_qk @ x + b   -> (b,n,512) f16 ----
  k_gemm<<<dim3(Nn / 128, 512 / 64, Bn), 256, 0, stream>>>(
      wqk, xh, qk_b, nullptr, nullptr, qkh, 512, Nn, 256, 0, 1);
  // ---- v_full = W_v @ x + b -> (b,c,n) f32 (pe conv) + (b,c,n) f16 (V frags) ----
  k_gemm<<<dim3(Nn / 128, 256 / 64, Bn), 256, 0, stream>>>(
      wv, xh, v_b, nullptr, vfull, vhh, 256, Nn, 256, 0, 0);
  // ---- pe = dwconv7x7(v_full) + pe_b ----
  k_dwconv7<<<cdiv(BCN, 256), 256, 0, stream>>>(vfull, pe_w, pe_b, pe, Bn, C, H, Wd);
  // ---- o = attention(q, k, v) ----
  k_flash<<<dim3(Nn / 16, NH, Bn), 32, 0, stream>>>(qkh, vhh, obuf, Nn);
  // ---- oph = f16(o + pe) -> (b,n,c) ----
  k_add_to_f16_t<<<cdiv(BCN, 256), 256, 0, stream>>>(obuf, pe, oph, C, Nn, BCN);
  // ---- x1 = x + W_proj @ (o+pe) + b  (f32 m-major + f16 k-major) ----
  k_gemm<<<dim3(Nn / 128, 256 / 64, Bn), 256, 0, stream>>>(
      wproj, oph, proj_b, x, x1, x1h, 256, Nn, 256, 0, 1);
  // ---- h = silu(W_fc1 @ x1 + b) -> (b,n,512) f16 ----
  k_gemm<<<dim3(Nn / 128, 512 / 64, Bn), 256, 0, stream>>>(
      wfc1, x1h, fc1_b, nullptr, nullptr, hbuf, 512, Nn, 256, 1, 1);
  // ---- out = x1 + W_fc2 @ h + b  (f32, final) ----
  k_gemm<<<dim3(Nn / 128, 256 / 64, Bn), 256, 0, stream>>>(
      wfc2, hbuf, fc2_b, x1, out, nullptr, 256, Nn, 512, 0, 0);
}
